// AdjacencyLearn_80221399155167
// MI455X (gfx1250) — matrix operand order repaired and checked by reading, not compiled, not run on hardware
//
#include <hip/hip_runtime.h>
#include <math.h>

// ---------------------------------------------------------------------------
// NRI (AdjacencyLearn) forward for MI455X / gfx1250, wave32 + WMMA.
// GEMMs run on v_wmma_f32_16x16x32_f16 with f32 accumulation. Weights are
// pre-packed once into f16 fragment order ([K/32][N/16][lane][16 halves]) so
// each lane feeds a whole B fragment with one 32B vector load; A fragments
// are float4 loads + cvt_pk; the epilogue is transposed through LDS for
// coalesced float4 stores. Scatters are closed-form gathers.
// ---------------------------------------------------------------------------

typedef __attribute__((ext_vector_type(16))) _Float16 v16h;
typedef __attribute__((ext_vector_type(8)))  float    v8f;

#define kV   25
#define kE   600      // V*(V-1)
#define kB   16       // N*M
#define kT   32
#define kC   3
#define kHE  128
#define kHD  64
#define kRN  400      // B*V node rows
#define kRE  9600     // B*E edge rows
#define kTAU 0.5f
#define kEPS 1e-10f
#define kDCY 0.1f
#define kBNE 1e-5f

#define ACT_NONE 0
#define ACT_ELU  1
#define ACT_TANH 2
#define ACT_RELU 3
#define ACT_SIG  4

#define ACC_STORE 0   // dst  = act(acc+bias)*scale
#define ACC_POST  1   // dst += act(acc+bias)*scale
#define ACC_PRE   2   // dst  = act(acc+bias+dst)*scale

__device__ __forceinline__ int edge_id(int s, int c) {
    return s * (kV - 1) + (c > s ? c - 1 : c);
}

__device__ __forceinline__ float act_apply(float v, int act) {
    switch (act) {
        case ACT_ELU:  return v > 0.f ? v : (__expf(v) - 1.f);
        case ACT_TANH: return tanhf(v);
        case ACT_RELU: return fmaxf(v, 0.f);
        case ACT_SIG:  return 1.f / (1.f + __expf(-v));
        default:       return v;
    }
}

// ---------------------------------------------------------------------------
// Weight pre-pack: f32 row-major W[K,N] -> f16 fragment order
// dst[((kt*(N/16)+nt)*32 + lane)*16 + e] = W[kt*32 + (lane>>4)*16 + e][nt*16 + (lane&15)]
// ---------------------------------------------------------------------------
__global__ void nri_pack_w(const float* __restrict__ Wsrc, _Float16* __restrict__ Wdst,
                           int K, int N) {
    int idx = blockIdx.x * blockDim.x + threadIdx.x;
    if (idx >= K * N) return;
    int e    = idx & 15;
    int lane = (idx >> 4) & 31;
    int tile = idx >> 9;              // kt * (N/16) + nt
    int nT   = N >> 4;
    int nt   = tile % nT;
    int kt   = tile / nT;
    int col  = nt * 16 + (lane & 15);
    int k    = kt * 32 + ((lane >> 4) << 4) + e;
    Wdst[idx] = (_Float16)Wsrc[(size_t)k * N + col];
}

// ---------------------------------------------------------------------------
// WMMA GEMM: C[M,N] (op)= act(A[M,K] @ W[K,N] + bias) * rowScale
// One wave per 16x64 tile, grid = (M/16, N/64). Wp is pre-packed f16.
// ---------------------------------------------------------------------------
__global__ __launch_bounds__(32)
void nri_wmma_gemm(const float* __restrict__ A, int lda,
                   const _Float16* __restrict__ Wp, int nTilesN,
                   const float* __restrict__ bias,
                   float* __restrict__ C, int ldc,
                   int K, int act, int accMode,
                   const float* __restrict__ rowScale)
{
    __shared__ float tile[16 * 64];
    const int lane = threadIdx.x;          // 0..31
    const int r16  = lane & 15;
    const int hi   = lane >> 4;
    const int arow = blockIdx.x * 16 + r16;
    const int kbA  = hi * 16;              // A: lanes16-31 carry K offset +16

    v8f acc[4] = {};
    const int nkt = K >> 5;

    for (int kt = 0; kt < nkt; ++kt) {
        const int k0 = kt << 5;
        // A fragment: one contiguous 16-float run -> 4x float4 + cvt
        const float* Ap = A + (size_t)arow * lda + k0;
        float4 a0 = *(const float4*)(Ap + kbA);
        float4 a1 = *(const float4*)(Ap + kbA + 4);
        float4 a2 = *(const float4*)(Ap + kbA + 8);
        float4 a3 = *(const float4*)(Ap + kbA + 12);
        v16h a;
        a[0]  = (_Float16)a0.x; a[1]  = (_Float16)a0.y;
        a[2]  = (_Float16)a0.z; a[3]  = (_Float16)a0.w;
        a[4]  = (_Float16)a1.x; a[5]  = (_Float16)a1.y;
        a[6]  = (_Float16)a1.z; a[7]  = (_Float16)a1.w;
        a[8]  = (_Float16)a2.x; a[9]  = (_Float16)a2.y;
        a[10] = (_Float16)a2.z; a[11] = (_Float16)a2.w;
        a[12] = (_Float16)a3.x; a[13] = (_Float16)a3.y;
        a[14] = (_Float16)a3.z; a[15] = (_Float16)a3.w;

        if (kt + 1 < nkt) {
            __builtin_prefetch(
                Wp + ((((size_t)(kt + 1) * nTilesN + blockIdx.y * 4) * 32 + lane) << 4), 0, 3);
        }
#pragma unroll
        for (int ct = 0; ct < 4; ++ct) {
            const int ctg = blockIdx.y * 4 + ct;
            // Whole B fragment for this lane: 16 contiguous halves (32B)
            const v16h* bp = (const v16h*)(Wp + ((((size_t)kt * nTilesN + ctg) * 32 + lane) << 4));
            v16h b = *bp;
            acc[ct] = __builtin_amdgcn_wmma_f32_16x16x32_f16(
                false, a, false, b, (short)0, acc[ct], false, false);
        }
    }

    // Stage accumulators to LDS (C/D layout: lane0-15 N=lane,M=v; lane16-31 N=lane-16,M=v+8)
#pragma unroll
    for (int ct = 0; ct < 4; ++ct)
#pragma unroll
        for (int v = 0; v < 8; ++v)
            tile[(v + hi * 8) * 64 + ct * 16 + r16] = acc[ct][v];
    __syncthreads();

    // Coalesced epilogue: lane handles half of one output row (32 floats)
    const int row = r16;
    const int gr  = blockIdx.x * 16 + row;
    const int gc0 = blockIdx.y * 64 + hi * 32;
    const float rs = rowScale ? rowScale[gr] : 1.f;
    float* dstRow = C + (size_t)gr * ldc + gc0;
#pragma unroll
    for (int q = 0; q < 8; ++q) {
        const float* tp = &tile[row * 64 + hi * 32 + q * 4];
        float o[4];
#pragma unroll
        for (int u = 0; u < 4; ++u) {
            float val = tp[u];
            if (bias) val += bias[gc0 + q * 4 + u];
            if (accMode == ACC_PRE) val += dstRow[q * 4 + u];
            val = act_apply(val, act);
            o[u] = val * rs;
        }
        if (accMode == ACC_POST) {
#pragma unroll
            for (int u = 0; u < 4; ++u) dstRow[q * 4 + u] += o[u];
        } else {
            *(float4*)(dstRow + q * 4) = make_float4(o[0], o[1], o[2], o[3]);
        }
    }
}

// ---------------------------------------------------------------------------
// x [N=8,C=3,T=32,V=25,M=2] -> xp [B=16,V,T,C] flattened [400, 96]
// ---------------------------------------------------------------------------
__global__ void nri_pack_x(const float* __restrict__ x, float* __restrict__ xp) {
    int idx = blockIdx.x * blockDim.x + threadIdx.x;
    if (idx >= kRN * kT * kC) return;
    int c = idx % kC; int t = (idx / kC) % kT;
    int v = (idx / (kC * kT)) % kV;
    int b = idx / (kC * kT * kV);
    int n = b >> 1, m = b & 1;
    xp[idx] = x[((((size_t)n * kC + c) * kT + t) * kV + v) * 2 + m];
}

// nodes [B,V,F] -> out rows [B,E], cols [recv(F) | send(F)], stride outStride
__global__ void nri_node2edge(const float* __restrict__ nodes, float* __restrict__ out,
                              int F, int outStride) {
    int idx = blockIdx.x * blockDim.x + threadIdx.x;
    int tot = kRE * 2 * F;
    if (idx >= tot) return;
    int j = idx % (2 * F);
    int row = idx / (2 * F);
    int b = row / kE, e = row % kE;
    int s = e / (kV - 1), rr = e % (kV - 1);
    int c = (rr < s) ? rr : rr + 1;          // receiver
    float val = (j < F) ? nodes[((size_t)b * kV + c) * F + j]
                        : nodes[((size_t)b * kV + s) * F + (j - F)];
    out[(size_t)row * outStride + j] = val;
}

// ecat[:, 256:384] = skip
__global__ void nri_copy_skip(const float* __restrict__ skip, float* __restrict__ ecat) {
    int idx = blockIdx.x * blockDim.x + threadIdx.x;
    if (idx >= kRE * kHE) return;
    int row = idx >> 7, j = idx & 127;
    ecat[(size_t)row * 384 + 256 + j] = skip[idx];
}

// out[b,v,h] = inv * sum_{s != v} eF[b, edge(s,v), h]
__global__ void nri_edge2node(const float* __restrict__ eF, float* __restrict__ out,
                              int F, float inv) {
    int idx = blockIdx.x * blockDim.x + threadIdx.x;
    if (idx >= kRN * F) return;
    int h = idx % F;
    int row = idx / F;
    int b = row / kV, v = row % kV;
    float s = 0.f;
#pragma unroll 4
    for (int snd = 0; snd < kV; ++snd) {
        if (snd == v) continue;
        s += eF[((size_t)b * kE + edge_id(snd, v)) * F + h];
    }
    out[idx] = s * inv;
}

// Column-wise batch-norm (training mode, biased var): bn[0:128]=scale, bn[128:256]=shift
__global__ void nri_bn_stats(const float* __restrict__ Y, int rows,
                             const float* __restrict__ g, const float* __restrict__ bb,
                             float* __restrict__ bn) {
    int j = threadIdx.x;
    float s = 0.f, ss = 0.f;
    for (int i = 0; i < rows; ++i) {
        float v = Y[(size_t)i * kHE + j];
        s += v; ss += v * v;
    }
    float mu  = s / rows;
    float var = ss / rows - mu * mu;
    float sc  = g[j] * rsqrtf(var + kBNE);
    bn[j]       = sc;
    bn[128 + j] = bb[j] - mu * sc;
}

__global__ void nri_bn_apply(const float* __restrict__ src, float* __restrict__ dst,
                             int n, const float* __restrict__ bn) {
    int idx = blockIdx.x * blockDim.x + threadIdx.x;
    if (idx >= n) return;
    int col = idx & 127;
    dst[idx] = src[idx] * bn[col] + bn[128 + col];
}

// logits[r,c] = e2[r,:]@fc_w[:,c] + fc_b[c]   (N=3 -> scalar kernel)
__global__ void nri_logits(const float* __restrict__ e2, const float* __restrict__ fw,
                           const float* __restrict__ fb, float* __restrict__ logits) {
    int idx = blockIdx.x * blockDim.x + threadIdx.x;
    if (idx >= kRE * kC) return;
    int c = idx % kC, row = idx / kC;
    float s = fb[c];
    for (int j = 0; j < kHE; ++j) s += e2[(size_t)row * kHE + j] * fw[j * kC + c];
    logits[idx] = s;
}

// Hard gumbel-softmax (ST) + prob; emits edges and escale[k-1] = edges[:,k]/norm
__global__ void nri_gumbel(const float* __restrict__ logits, const float* __restrict__ u,
                           float* __restrict__ probOut, float* __restrict__ edges,
                           float* __restrict__ escale) {
    int r = blockIdx.x * blockDim.x + threadIdx.x;
    if (r >= kRE) return;
    float l[kC], z[kC], ys[kC], pr[kC];
    float zmax = -1e30f, lmax = -1e30f;
    for (int c = 0; c < kC; ++c) {
        l[c] = logits[r * kC + c];
        float g = -logf(kEPS - logf(u[r * kC + c] + kEPS));
        z[c] = (l[c] + g) / kTAU;
        zmax = fmaxf(zmax, z[c]); lmax = fmaxf(lmax, l[c]);
    }
    float zs = 0.f, ls = 0.f;
    for (int c = 0; c < kC; ++c) { ys[c] = __expf(z[c] - zmax); zs += ys[c];
                                   pr[c] = __expf(l[c] - lmax); ls += pr[c]; }
    int amax = 0; float best = -1e30f;
    for (int c = 0; c < kC; ++c) {
        ys[c] /= zs; pr[c] /= ls;
        if (ys[c] > best) { best = ys[c]; amax = c; }
    }
    for (int c = 0; c < kC; ++c) {
        float yh = (c == amax) ? 1.f : 0.f;
        float e  = (yh - ys[c]) + ys[c];     // straight-through forward
        edges[r * kC + c]   = e;
        probOut[r * kC + c] = pr[c];
        if (c >= 1) escale[(c - 1) * kRE + r] = e * 0.5f;  // / (NUM_TYPES-1)
    }
}

__global__ void nri_zero(float* __restrict__ p, int n) {
    int idx = blockIdx.x * blockDim.x + threadIdx.x;
    if (idx < n) p[idx] = 0.f;
}

// GRU input projections (K=3, scalar): pre_{r,i,n}[row,h] = ins@in_*_w + in_*_b
__global__ void nri_gru_in(const float* __restrict__ xp, int t,
                           const float* __restrict__ wr, const float* __restrict__ br,
                           const float* __restrict__ wi, const float* __restrict__ bi,
                           const float* __restrict__ wn, const float* __restrict__ bnn,
                           float* __restrict__ pr, float* __restrict__ pi,
                           float* __restrict__ pn) {
    int idx = blockIdx.x * blockDim.x + threadIdx.x;
    if (idx >= kRN * kHD) return;
    int h = idx % kHD, row = idx / kHD;
    const float* ins = xp + (size_t)row * (kT * kC) + t * kC;
    float r = br[h], i = bi[h], n = bnn[h];
    for (int c = 0; c < kC; ++c) {
        float v = ins[c];
        r += v * wr[c * kHD + h];
        i += v * wi[c * kHD + h];
        n += v * wn[c * kHD + h];
    }
    pr[idx] = r; pi[idx] = i; pn[idx] = n;
}

// hidden = (1-i)*tanh(pre_n + r*gn) + i*hidden  (r,i already sigmoided by GEMM)
__global__ void nri_gru_update(const float* __restrict__ r, const float* __restrict__ i,
                               const float* __restrict__ pn, const float* __restrict__ gn,
                               float* __restrict__ hidden) {
    int idx = blockIdx.x * blockDim.x + threadIdx.x;
    if (idx >= kRN * kHD) return;
    float nb = tanhf(pn[idx] + r[idx] * gn[idx]);
    hidden[idx] = (1.f - i[idx]) * nb + i[idx] * hidden[idx];
}

// pred[b,v,t,c] = ins + p2@out3_w[:,c] + out3_b[c]  -> d_out (outputs slab)
__global__ void nri_pred(const float* __restrict__ xp, int t,
                         const float* __restrict__ p2, const float* __restrict__ w3,
                         const float* __restrict__ b3, float* __restrict__ out) {
    int idx = blockIdx.x * blockDim.x + threadIdx.x;
    if (idx >= kRN * kC) return;
    int c = idx % kC, row = idx / kC;
    float s = xp[(size_t)row * (kT * kC) + t * kC + c] + b3[c];
    for (int j = 0; j < kHD; ++j) s += p2[(size_t)row * kHD + j] * w3[j * kC + c];
    out[((size_t)row * (kT - 1) + t) * kC + c] = s;
}

// Per-sample normalized adjacency: A = (eye + edges_offdiag), column-normalized.
__global__ void nri_adj(const float* __restrict__ edges, float* __restrict__ out) {
    int b = blockIdx.x, k = blockIdx.y, j = threadIdx.x;
    if (j >= kV) return;
    float colsum = 0.f;
    for (int i = 0; i < kV; ++i) {
        float a = (i == j) ? 1.f : edges[((size_t)b * kE + edge_id(i, j)) * kC + (k + 1)];
        colsum += a;
    }
    float d = (1.f / colsum + kEPS) * kDCY;
    for (int i = 0; i < kV; ++i) {
        float a = (i == j) ? 1.f : edges[((size_t)b * kE + edge_id(i, j)) * kC + (k + 1)];
        out[(((size_t)b * 2 + k) * kV + i) * kV + j] = a * d;
    }
}

// ---------------------------------------------------------------------------
// Host driver
// ---------------------------------------------------------------------------
extern "C" void kernel_launch(void* const* d_in, const int* in_sizes, int n_in,
                              void* d_out, int out_size, void* d_ws, size_t ws_size,
                              hipStream_t stream) {
    (void)in_sizes; (void)out_size; (void)ws_size;

    // Inputs in setup_inputs() recursive dict (insertion) order.
    const float* P[64];
    for (int i = 0; i < n_in && i < 64; ++i) P[i] = (const float*)d_in[i];
    enum {
        IN_X = 0, IN_GU = 1,
        M1 = 2,   // mlp1: w1,b1,w2,b2,g,bb
        M2 = 8, M3 = 14, M4 = 20,
        FC_W = 26, FC_B = 27,
        MSG_W1 = 28, MSG_B1 = 29, MSG_W2 = 30, MSG_B2 = 31,
        HID_R = 32, HID_I = 33, HID_N = 34,
        INR_W = 35, INR_B = 36, INI_W = 37, INI_B = 38, INN_W = 39, INN_B = 40,
        O1_W = 41, O1_B = 42, O2_W = 43, O2_B = 44, O3_W = 45, O3_B = 46
    };

    // Output slabs: outputs [16,25,31,3] | prob [16,600,3] | A [16,2,25,25]
    float* outPred = (float*)d_out;
    float* outProb = outPred + kB * kV * (kT - 1) * kC;   // +37200
    float* outA    = outProb + kB * kE * kC;              // +28800

    // Workspace: float arena then f16 packed-weight arena.
    float* W = (float*)d_ws;
    size_t off = 0;
    auto alloc = [&](size_t n) { float* p = W + off; off += n; return p; };
    float* xp    = alloc((size_t)kRN * kT * kC);
    float* big0  = alloc((size_t)kRE * 384);         // eh/ecat; decoder pre
    float* big1  = alloc((size_t)kRE * kHE);         // Y1; decoder m1; mlp4 result
    float* big2  = alloc((size_t)kRE * kHE);         // Y2; decoder all_msgs
    float* hbuf  = alloc((size_t)kRN * kHE);
    float* skip  = alloc((size_t)kRE * kHE);
    float* nin   = alloc((size_t)kRN * kHE);
    float* nfeat = alloc((size_t)kRN * kHE);
    float* lgt   = alloc((size_t)kRE * kC);
    float* edg   = alloc((size_t)kRE * kC);
    float* escl  = alloc((size_t)2 * kRE);
    float* bnb   = alloc(256);
    float* hid   = alloc((size_t)kRN * kHD);
    float* agg   = alloc((size_t)kRN * kHD);
    float* preR  = alloc((size_t)kRN * kHD);
    float* preI  = alloc((size_t)kRN * kHD);
    float* preN  = alloc((size_t)kRN * kHD);
    float* gnb   = alloc((size_t)kRN * kHD);
    float* p1b   = alloc((size_t)kRN * kHD);
    float* p2b   = alloc((size_t)kRN * kHD);

    off = (off + 15) & ~(size_t)15;                  // 64B-align f16 arena
    _Float16* Hbase = (_Float16*)(W + off);
    size_t hoff = 0;
    auto halloc = [&](size_t n) { _Float16* p = Hbase + hoff; hoff += n; return p; };

    auto grid1 = [](int n) { return dim3((n + 255) / 256); };
    auto packw = [&](const float* src, _Float16* dst, int K, int N) {
        nri_pack_w<<<grid1(K * N), 256, 0, stream>>>(src, dst, K, N);
    };

    // Pre-pack every WMMA weight once (f32 -> f16 fragment order).
    _Float16* w_m1w1 = halloc(96 * 128);  packw(P[M1],     w_m1w1, 96,  128);
    _Float16* w_m1w2 = halloc(128 * 128); packw(P[M1 + 2], w_m1w2, 128, 128);
    _Float16* w_m2w1 = halloc(256 * 128); packw(P[M2],     w_m2w1, 256, 128);
    _Float16* w_m2w2 = halloc(128 * 128); packw(P[M2 + 2], w_m2w2, 128, 128);
    _Float16* w_m3w1 = halloc(128 * 128); packw(P[M3],     w_m3w1, 128, 128);
    _Float16* w_m3w2 = halloc(128 * 128); packw(P[M3 + 2], w_m3w2, 128, 128);
    _Float16* w_m4w1 = halloc(384 * 128); packw(P[M4],     w_m4w1, 384, 128);
    _Float16* w_m4w2 = halloc(128 * 128); packw(P[M4 + 2], w_m4w2, 128, 128);
    _Float16* w_mw1[3] = {nullptr, nullptr, nullptr};
    _Float16* w_mw2[3] = {nullptr, nullptr, nullptr};
    for (int k = 1; k < kC; ++k) {
        w_mw1[k] = halloc(128 * 64); packw(P[MSG_W1] + (size_t)k * 128 * 64, w_mw1[k], 128, 64);
        w_mw2[k] = halloc(64 * 64);  packw(P[MSG_W2] + (size_t)k * 64 * 64,  w_mw2[k], 64,  64);
    }
    _Float16* w_hr = halloc(64 * 64); packw(P[HID_R], w_hr, 64, 64);
    _Float16* w_hi = halloc(64 * 64); packw(P[HID_I], w_hi, 64, 64);
    _Float16* w_hn = halloc(64 * 64); packw(P[HID_N], w_hn, 64, 64);
    _Float16* w_o1 = halloc(64 * 64); packw(P[O1_W],  w_o1, 64, 64);
    _Float16* w_o2 = halloc(64 * 64); packw(P[O2_W],  w_o2, 64, 64);

    auto gemm = [&](const float* A, int lda, const _Float16* Wp, const float* bias,
                    float* C, int ldc, int M, int N, int K, int act, int accMode,
                    const float* rowScale) {
        dim3 g(M / 16, N / 64);
        nri_wmma_gemm<<<g, 32, 0, stream>>>(A, lda, Wp, N / 16, bias, C, ldc,
                                            K, act, accMode, rowScale);
    };
    // _mlp: fc1->elu->fc2->elu->BN(train)
    auto run_mlp = [&](const float* X, int rows, int K1, const _Float16* w1p,
                       const _Float16* w2p, int pbase, float* dst) {
        const float *b1 = P[pbase + 1], *b2 = P[pbase + 3],
                    *g = P[pbase + 4], *bb = P[pbase + 5];
        gemm(X, K1, w1p, b1, big1, kHE, rows, kHE, K1, ACT_ELU, ACC_STORE, nullptr);
        gemm(big1, kHE, w2p, b2, big2, kHE, rows, kHE, kHE, ACT_ELU, ACC_STORE, nullptr);
        nri_bn_stats<<<1, 128, 0, stream>>>(big2, rows, g, bb, bnb);
        nri_bn_apply<<<grid1(rows * kHE), 256, 0, stream>>>(big2, dst, rows * kHE, bnb);
    };

    // ---------------- encoder ----------------
    nri_pack_x<<<grid1(kRN * kT * kC), 256, 0, stream>>>(P[IN_X], xp);
    run_mlp(xp, kRN, kT * kC, w_m1w1, w_m1w2, M1, hbuf);                   // h [400,128]
    nri_node2edge<<<grid1(kRE * 256), 256, 0, stream>>>(hbuf, big0, kHE, 256);
    run_mlp(big0, kRE, 256, w_m2w1, w_m2w2, M2, skip);                     // e/skip
    nri_edge2node<<<grid1(kRN * kHE), 256, 0, stream>>>(skip, nin, kHE, 1.f / kV);
    run_mlp(nin, kRN, kHE, w_m3w1, w_m3w2, M3, nfeat);                     // n [400,128]
    nri_node2edge<<<grid1(kRE * 256), 256, 0, stream>>>(nfeat, big0, kHE, 384);
    nri_copy_skip<<<grid1(kRE * kHE), 256, 0, stream>>>(skip, big0);
    run_mlp(big0, kRE, 384, w_m4w1, w_m4w2, M4, big1);                     // e2 [9600,128]
    nri_logits<<<grid1(kRE * kC), 256, 0, stream>>>(big1, P[FC_W], P[FC_B], lgt);

    // ---------------- gumbel-softmax ----------------
    nri_gumbel<<<grid1(kRE), 256, 0, stream>>>(lgt, P[IN_GU], outProb, edg, escl);

    // ---------------- decoder (31 steps) ----------------
    nri_zero<<<grid1(kRN * kHD), 256, 0, stream>>>(hid, kRN * kHD);
    for (int t = 0; t < kT - 1; ++t) {
        nri_node2edge<<<grid1(kRE * 2 * kHD), 256, 0, stream>>>(hid, big0, kHD, 2 * kHD);
        for (int k = 1; k < kC; ++k) {      // skip first edge type
            gemm(big0, 2 * kHD, w_mw1[k], P[MSG_B1] + k * kHD, big1, kHD,
                 kRE, kHD, 2 * kHD, ACT_TANH, ACC_STORE, nullptr);
            gemm(big1, kHD, w_mw2[k], P[MSG_B2] + k * kHD, big2, kHD,
                 kRE, kHD, kHD, ACT_TANH, (k == 1) ? ACC_STORE : ACC_POST,
                 escl + (size_t)(k - 1) * kRE);
        }
        nri_edge2node<<<grid1(kRN * kHD), 256, 0, stream>>>(big2, agg, kHD, 1.f / kC);
        nri_gru_in<<<grid1(kRN * kHD), 256, 0, stream>>>(
            xp, t, P[INR_W], P[INR_B], P[INI_W], P[INI_B], P[INN_W], P[INN_B],
            preR, preI, preN);
        gemm(agg, kHD, w_hr, nullptr, preR, kHD, kRN, kHD, kHD, ACT_SIG, ACC_PRE, nullptr);
        gemm(agg, kHD, w_hi, nullptr, preI, kHD, kRN, kHD, kHD, ACT_SIG, ACC_PRE, nullptr);
        gemm(agg, kHD, w_hn, nullptr, gnb,  kHD, kRN, kHD, kHD, ACT_NONE, ACC_STORE, nullptr);
        nri_gru_update<<<grid1(kRN * kHD), 256, 0, stream>>>(preR, preI, preN, gnb, hid);
        gemm(hid, kHD, w_o1, P[O1_B], p1b, kHD, kRN, kHD, kHD, ACT_RELU, ACC_STORE, nullptr);
        gemm(p1b, kHD, w_o2, P[O2_B], p2b, kHD, kRN, kHD, kHD, ACT_RELU, ACC_STORE, nullptr);
        nri_pred<<<grid1(kRN * kC), 256, 0, stream>>>(xp, t, p2b, P[O3_W], P[O3_B], outPred);
    }

    // ---------------- adjacency ----------------
    nri_adj<<<dim3(kB, kC - 1), 32, 0, stream>>>(edg, outA);
}
